// FastAttention_7851200217500
// MI455X (gfx1250) — matrix operand order repaired and checked by reading, not compiled, not run on hardware
//
#include <hip/hip_runtime.h>

typedef __attribute__((ext_vector_type(16))) _Float16 v16h;
typedef __attribute__((ext_vector_type(8)))  _Float16 v8h;
typedef __attribute__((ext_vector_type(8)))  float    v8f;

#define BH   32      // b*h
#define SEQ  4096
#define DH   64
#define NF   256
#define DN   0.35355339059327373f   // 64^-0.25
#define RATIO 0.0625f               // 256^-0.5
#define KEPS 1e-4f

#define LDP 264      // padded row length (halves/floats) for 256-wide tiles
#define LDV 72       // padded f16 row length for 64-wide x tiles
#define LDK 40       // padded f16 row length for 32-wide (K=n32) tiles
#define SPLITB 8

// workspace layout (floats); ctx stored TRANSPOSED: [bh][e][j]
#define CTX_OFF  0
#define KSUM_OFF (BH * NF * DH)          // 524288
#define BMAX_OFF (KSUM_OFF + BH * NF)    // 532480
#define GMAX_OFF (BMAX_OFF + BH * 128)   // 536576

// ---- WMMA fragment helpers (wave32, 16x16x32 f16) -------------------------
// A (16x32): lane m=L%16, slots 0..15 -> K = hi+0..15 (contiguous),
// hi = (L>=16)?16:0.  Row-major source => two 16B contiguous LDS loads.
__device__ __forceinline__ v16h load_a_frag(const _Float16* s, int ld, int m0, int k0, int lane) {
  const _Float16* p = s + (m0 + (lane & 15)) * ld + k0 + ((lane >= 16) ? 16 : 0);
  v8h lo = *(const v8h*)p;
  v8h hi = *(const v8h*)(p + 8);
  return __builtin_shufflevector(lo, hi, 0,1,2,3,4,5,6,7,8,9,10,11,12,13,14,15);
}

// B (32x16): lane n=L%16, K = (L>=16?16:0) + e (contiguous 16).
// n-major source ([n][k] row-major) => one 32B contiguous LDS load.
__device__ __forceinline__ v16h load_b_frag(const _Float16* s, int ld, int n0, int k0, int lane) {
  const _Float16* p = s + (n0 + (lane & 15)) * ld + k0 + ((lane >= 16) ? 16 : 0);
  return *(const v16h*)p;
}

// B fragment straight from a row-major f32 global matrix (n-major), with cvt.
__device__ __forceinline__ v16h load_b_global_f32(const float* g, int ldg, int n0, int k0, int lane) {
  const float* p = g + (size_t)(n0 + (lane & 15)) * ldg + k0 + ((lane >= 16) ? 16 : 0);
  v16h f;
  #pragma unroll
  for (int e = 0; e < 16; ++e) f[e] = (_Float16)p[e];
  return f;
}

__device__ __forceinline__ v8f wmma16(v16h a, v16h b, v8f c) {
  return __builtin_amdgcn_wmma_f32_16x16x32_f16(false, a, false, b, (short)0, c, false, false);
}

// Per-wave projection B fragments (held in registers): jl in {0,1} -> j-tile 2w+jl,
// ks in {0,1} -> K step.  proj is [256][64] f32 row-major == n-major for B.
__device__ __forceinline__ void load_pfrag(const float* proj, v16h bf[4], int w, int lane) {
  #pragma unroll
  for (int jl = 0; jl < 2; ++jl)
    #pragma unroll
    for (int ks = 0; ks < 2; ++ks)
      bf[jl * 2 + ks] = load_b_global_f32(proj, DH, (2 * w + jl) * 16, ks * 32, lane);
}

// dd tile: [32 rows x 256 feat] = (DN*x)[32x64] @ projT.  Wave w owns 4 output
// tiles: i -> rt = i&1 (row tile), jt = 2w + (i>>1) (feature tile).
__device__ __forceinline__ void proj_mm(const _Float16 (*xt)[LDV], const v16h bf[4],
                                        v8f dd[4], int lane) {
  #pragma unroll
  for (int i = 0; i < 4; ++i) {
    int rt = i & 1, jl = i >> 1;
    v8f acc = {};
    #pragma unroll
    for (int ks = 0; ks < 2; ++ks) {
      v16h a = load_a_frag(&xt[0][0], LDV, rt * 16, ks * 32, lane);
      acc = wmma16(a, bf[jl * 2 + ks], acc);
    }
    dd[i] = acc;
  }
}

// ---- tile loaders ----------------------------------------------------------
__device__ __forceinline__ void load_x32(const float* xb, int row0,
                                         _Float16 (*xt)[LDV],
                                         float* diag, float (*pd)[8], bool do_diag) {
  int t = threadIdx.x, r = t >> 3, c = (t & 7) * 8;
  const float* p = xb + (size_t)(row0 + r) * DH + c;
  float s = 0.f;
  #pragma unroll
  for (int i = 0; i < 8; ++i) {
    float v = p[i];
    s += v * v;
    xt[r][c + i] = (_Float16)(DN * v);
  }
  if (do_diag) pd[r][t & 7] = s;
  __syncthreads();
  if (do_diag && t < 32) {
    float d = 0.f;
    #pragma unroll
    for (int i = 0; i < 8; ++i) d += pd[t][i];
    diag[t] = 0.0625f * d;
  }
}

// v tile transposed into e-major LDS: vtT[e][n32]
__device__ __forceinline__ void load_vT32(const float* vb, int row0, _Float16 (*vtT)[LDK]) {
  int t = threadIdx.x, r = t >> 3, c = (t & 7) * 8;
  const float* p = vb + (size_t)(row0 + r) * DH + c;
  #pragma unroll
  for (int i = 0; i < 8; ++i) vtT[c + i][r] = (_Float16)p[i];
}

// ---- kernels ---------------------------------------------------------------
__global__ void perf_zero(float* __restrict__ ws, int n) {
  int i = blockIdx.x * blockDim.x + threadIdx.x;
  if (i < n) ws[i] = 0.f;
}

// Kernel A: WMMA data_dash for k tiles, block max.
__global__ __launch_bounds__(256) void perf_kmax(const float* __restrict__ k,
                                                 const float* __restrict__ proj,
                                                 float* __restrict__ bmax) {
  __shared__ __align__(16) _Float16 xt[32][LDV];
  __shared__ float red[256];
  int t = threadIdx.x, w = t >> 5, lane = t & 31;
  int bh = blockIdx.x >> 7, nt = blockIdx.x & 127;
  const float* kb = k + (size_t)bh * SEQ * DH;
  v16h pfrag[4];
  load_pfrag(proj, pfrag, w, lane);
  load_x32(kb, nt * 32, xt, nullptr, nullptr, false);
  __syncthreads();
  v8f dd[4];
  proj_mm(xt, pfrag, dd, lane);
  float m = -3.0e38f;
  #pragma unroll
  for (int i = 0; i < 4; ++i)
    #pragma unroll
    for (int r = 0; r < 8; ++r) m = fmaxf(m, dd[i][r]);
  red[t] = m;
  __syncthreads();
  for (int s = 128; s > 0; s >>= 1) {
    if (t < s) red[t] = fmaxf(red[t], red[t + s]);
    __syncthreads();
  }
  if (t == 0) bmax[blockIdx.x] = red[0];
}

__global__ __launch_bounds__(256) void perf_gmax(const float* __restrict__ bmax,
                                                 float* __restrict__ gmax) {
  __shared__ float red[256];
  int t = threadIdx.x;
  float m = -3.0e38f;
  for (int i = t; i < BH * 128; i += 256) m = fmaxf(m, bmax[i]);
  red[t] = m;
  __syncthreads();
  for (int s = 128; s > 0; s >>= 1) {
    if (t < s) red[t] = fmaxf(red[t], red[t + s]);
    __syncthreads();
  }
  if (t == 0) gmax[0] = red[0];
}

// Kernel B: kp tiles on the fly (global max); context accumulated in per-wave
// WMMA accumulators; flush via f32 atomics into transposed ctx [bh][e][j].
__global__ __launch_bounds__(256) void perf_kctx(const float* __restrict__ k,
                                                 const float* __restrict__ v,
                                                 const float* __restrict__ proj,
                                                 const float* __restrict__ gmaxp,
                                                 float* __restrict__ ctx,
                                                 float* __restrict__ ksumg) {
  __shared__ __align__(16) _Float16 kpT[NF][LDK];   // kp transposed: [j][n32]
  __shared__ __align__(16) _Float16 xt[32][LDV];
  __shared__ __align__(16) _Float16 vtT[DH][LDK];   // v transposed: [e][n32]
  __shared__ float ksum[NF];
  __shared__ float diag[32];
  __shared__ float pd[32][8];
  int t = threadIdx.x, w = t >> 5, lane = t & 31;
  int bh = blockIdx.x / SPLITB, sp = blockIdx.x % SPLITB;
  const float* kb = k + (size_t)bh * SEQ * DH;
  const float* vb = v + (size_t)bh * SEQ * DH;
  float gmax = gmaxp[0];
  ksum[t] = 0.f;
  v16h pfrag[4];
  load_pfrag(proj, pfrag, w, lane);
  v8f cacc[8] = {};
  __syncthreads();
  const int rows = SEQ / SPLITB;
  for (int it = 0; it < rows / 32; ++it) {
    int r0 = sp * rows + it * 32;
    load_x32(kb, r0, xt, diag, pd, true);
    load_vT32(vb, r0, vtT);
    __syncthreads();
    v8f dd[4];
    proj_mm(xt, pfrag, dd, lane);
    // exp with global max -> packed 16B stores into kpT
    int hi = (lane < 16) ? 0 : 8;
    #pragma unroll
    for (int i = 0; i < 4; ++i) {
      int rt = i & 1, jt = 2 * w + (i >> 1);
      int jc = jt * 16 + (lane & 15);
      v8h pk;
      #pragma unroll
      for (int r = 0; r < 8; ++r) {
        int row = rt * 16 + hi + r;
        pk[r] = (_Float16)(RATIO * (__expf(dd[i][r] - diag[row] - gmax) + KEPS));
      }
      *(v8h*)&kpT[jc][rt * 16 + hi] = pk;
    }
    __syncthreads();
    // k_sum: thread t owns feature j=t, contiguous 32-half row of kpT
    {
      float s = 0.f;
      #pragma unroll 8
      for (int r = 0; r < 32; ++r) s += (float)kpT[t][r];
      ksum[t] += s;
    }
    // context += kp^T @ v : wave w owns jt {2w,2w+1} x et 0..3
    v16h af[2], bf[4];
    af[0] = load_a_frag(&kpT[0][0], LDK, (2 * w + 0) * 16, 0, lane);
    af[1] = load_a_frag(&kpT[0][0], LDK, (2 * w + 1) * 16, 0, lane);
    #pragma unroll
    for (int et = 0; et < 4; ++et) bf[et] = load_b_frag(&vtT[0][0], LDK, et * 16, 0, lane);
    #pragma unroll
    for (int c = 0; c < 8; ++c) cacc[c] = wmma16(af[c >> 2], bf[c & 3], cacc[c]);
    __syncthreads();
  }
  atomicAdd(&ksumg[bh * NF + t], ksum[t]);
  int hi = (lane < 16) ? 0 : 8;
  #pragma unroll
  for (int c = 0; c < 8; ++c) {
    int jt = 2 * w + (c >> 2), et = c & 3;
    int e = et * 16 + (lane & 15);
    #pragma unroll
    for (int r = 0; r < 8; ++r) {
      int j = jt * 16 + hi + r;
      atomicAdd(&ctx[((size_t)bh * DH + e) * NF + j], cacc[c][r]);
    }
  }
}

// Kernel C: qp tiles (per-row max), d_inv, out = (qp@context)*d_inv.
__global__ __launch_bounds__(256) void perf_out(const float* __restrict__ q,
                                                const float* __restrict__ proj,
                                                const float* __restrict__ ctx,
                                                const float* __restrict__ ksumg,
                                                float* __restrict__ out) {
  // union region U: dd f32 [32][264] (33792B) -> ctxT f16 [64][264] (33792B)
  __shared__ __align__(16) unsigned char smem[33792 + 16896 + 4608 + 1024 + 1024 + 128 + 128 + 128];
  unsigned char* p = smem;
  float    (*U_dd)[LDP]  = (float(*)[LDP])p;
  _Float16 (*U_ctx)[LDP] = (_Float16(*)[LDP])p;
  p += 33792;
  _Float16 (*qp)[LDP] = (_Float16(*)[LDP])p; p += 16896;
  _Float16 (*xt)[LDV] = (_Float16(*)[LDV])p; p += 4608;
  float* ksum = (float*)p;           p += 1024;
  float (*pred)[8] = (float(*)[8])p; p += 1024;
  float* diag = (float*)p;           p += 128;
  float* rmax = (float*)p;           p += 128;
  float* dinv = (float*)p;

  int t = threadIdx.x, w = t >> 5, lane = t & 31;
  int bh = blockIdx.x >> 7, nt = blockIdx.x & 127;
  const float* qb = q + (size_t)bh * SEQ * DH;

  v16h pfrag[4];
  load_pfrag(proj, pfrag, w, lane);
  load_x32(qb, nt * 32, xt, diag, pred, true);
  __syncthreads();
  v8f dd[4];
  proj_mm(xt, pfrag, dd, lane);
  // stash raw data_dash (row-major for contiguous max/exp scans)
  int hi = (lane < 16) ? 0 : 8;
  #pragma unroll
  for (int i = 0; i < 4; ++i) {
    int rt = i & 1, jt = 2 * w + (i >> 1);
    int jc = jt * 16 + (lane & 15);
    #pragma unroll
    for (int r = 0; r < 8; ++r) U_dd[rt * 16 + hi + r][jc] = dd[i][r];
  }
  __syncthreads();
  // per-row max over 256 features
  {
    int r = t >> 3, j0 = (t & 7) * 32;
    float m = -3.0e38f;
    #pragma unroll 8
    for (int j = 0; j < 32; ++j) m = fmaxf(m, U_dd[r][j0 + j]);
    pred[r][t & 7] = m;
  }
  __syncthreads();
  if (t < 32) {
    float m = -3.0e38f;
    #pragma unroll
    for (int i = 0; i < 8; ++i) m = fmaxf(m, pred[t][i]);
    rmax[t] = m;
  }
  __syncthreads();
  // exp -> qp (f16, packed 16B stores)
  {
    int r = t >> 3, j0 = (t & 7) * 32;
    float dg = diag[r], mx = rmax[r];
    #pragma unroll
    for (int jb = 0; jb < 32; jb += 8) {
      v8h qv;
      #pragma unroll
      for (int u = 0; u < 8; ++u)
        qv[u] = (_Float16)(RATIO * (__expf(U_dd[r][j0 + jb + u] - dg - mx) + KEPS));
      *(v8h*)&qp[r][j0 + jb] = qv;
    }
  }
  __syncthreads();   // dd consumed, U region free
  ksum[t] = ksumg[bh * NF + t];
  {
    int e = t >> 2, j0 = (t & 3) * 64;   // ctx ws is [bh][e][j]
    const float* cb = ctx + ((size_t)bh * DH + e) * NF + j0;
    #pragma unroll 8
    for (int i = 0; i < 64; ++i) U_ctx[e][j0 + i] = (_Float16)cb[i];
  }
  __syncthreads();
  // d = qp . k_sum per row
  {
    int r = t >> 3, j0 = (t & 7) * 32;
    float s = 0.f;
    #pragma unroll 8
    for (int j = 0; j < 32; ++j) s += (float)qp[r][j0 + j] * ksum[j0 + j];
    pred[r][t & 7] = s;
  }
  __syncthreads();
  if (t < 32) {
    float s = 0.f;
    #pragma unroll
    for (int i = 0; i < 8; ++i) s += pred[t][i];
    dinv[t] = 1.f / s;
  }
  __syncthreads();
  // out tile: wave w -> rt = w&1 (rows), et = w>>1 (dim-head cols), K=256
  int rt = w & 1, et = w >> 1;
  v8f acc = {};
  #pragma unroll
  for (int ks = 0; ks < 8; ++ks) {
    v16h a = load_a_frag(&qp[0][0], LDP, rt * 16, ks * 32, lane);
    v16h b = load_b_frag(&U_ctx[0][0], LDP, et * 16, ks * 32, lane);
    acc = wmma16(a, b, acc);
  }
  int e = et * 16 + (lane & 15);
  float* ob = out + ((size_t)bh * SEQ + (size_t)nt * 32) * DH;
  #pragma unroll
  for (int r = 0; r < 8; ++r) {
    int row = rt * 16 + hi + r;
    ob[(size_t)row * DH + e] = acc[r] * dinv[row];
  }
}

extern "C" void kernel_launch(void* const* d_in, const int* in_sizes, int n_in,
                              void* d_out, int out_size, void* d_ws, size_t ws_size,
                              hipStream_t stream) {
  (void)in_sizes; (void)n_in; (void)out_size; (void)ws_size;
  const float* q    = (const float*)d_in[0];
  const float* k    = (const float*)d_in[1];
  const float* v    = (const float*)d_in[2];
  const float* proj = (const float*)d_in[3];
  float* out = (float*)d_out;
  float* ws  = (float*)d_ws;

  int nz = BMAX_OFF;  // zero context + k_sum accumulators
  perf_zero<<<(nz + 255) / 256, 256, 0, stream>>>(ws, nz);
  perf_kmax<<<BH * 128, 256, 0, stream>>>(k, proj, ws + BMAX_OFF);
  perf_gmax<<<1, 256, 0, stream>>>(ws + BMAX_OFF, ws + GMAX_OFF);
  perf_kctx<<<BH * SPLITB, 256, 0, stream>>>(k, v, proj, ws + GMAX_OFF,
                                             ws + CTX_OFF, ws + KSUM_OFF);
  perf_out<<<BH * 128, 256, 0, stream>>>(q, proj, ws + CTX_OFF, ws + KSUM_OFF, out);
}
